// Block_7868380086724
// MI455X (gfx1250) — matrix operand order repaired and checked by reading, not compiled, not run on hardware
//
#include <hip/hip_runtime.h>
#include <hip/hip_bf16.h>
#include <math.h>

// ---------------- problem constants ----------------
#define N_PTS 200000
#define CDIM  96
#define KOFF  343
#define HID   384
#define TM    128
#define NT    ((N_PTS + TM - 1) / TM)   // 1563 tiles

// ---------------- vector types for WMMA ----------------
typedef __bf16        v16bf __attribute__((ext_vector_type(16)));
typedef float         v8f   __attribute__((ext_vector_type(8)));
typedef unsigned int  v4u   __attribute__((ext_vector_type(4)));

union BFrag { v4u q[2]; v16bf v; };

__device__ __forceinline__ unsigned short f2bf(float f) {
    union { float f; unsigned u; } cv; cv.f = f;
    unsigned r = cv.u + 0x7FFFu + ((cv.u >> 16) & 1u);   // round-to-nearest-even
    return (unsigned short)(r >> 16);
}

// two 16B chunks -> one 16-element bf16 fragment
__device__ __forceinline__ v16bf make_frag(const unsigned short* p0, const unsigned short* p1) {
    BFrag f;
    f.q[0] = *(const v4u*)p0;
    f.q[1] = *(const v4u*)p1;
    return f.v;
}

// ---------------- CDNA5 async copy to LDS ----------------
// per-lane: LDS[ldsOff] = MEM[gaddr] (16 bytes), tracked by ASYNCcnt
__device__ __forceinline__ void async_b128(unsigned ldsOff, const void* gaddr) {
    asm volatile("global_load_async_to_lds_b128 %0, %1, off"
                 :: "v"(ldsOff), "v"(gaddr) : "memory");
}
// wait until ASYNCcnt <= N (async loads complete in order)
template <int N>
__device__ __forceinline__ void wait_async_le() {
    asm volatile("s_wait_asynccnt %0" :: "n"(N) : "memory");
}

// ---------------- LDS layout ----------------
#define SAS 104                       // A-tile row stride (halfs), padded vs bank conflicts
#define SWS 104                       // W-stage row stride (halfs)
#define SXS 100                       // conv-out row stride (floats)
#define SHS 392                       // hidden row stride (halfs)
#define SA_BYTES (TM * SAS * 2)       // 26,624  (one A buffer)
#define SW_BYTES (96 * SWS * 2)       // 19,968  (one W buffer)
#define SX_BYTES (TM * SXS * 4)       // 51,200
#define SH_BYTES (TM * SHS * 2)       // 100,352
// conv phase: [sA0][sA1][sW0][sW1] = 93,184
// post phase: [LN out @0][sX / sH @ 26,624]  -> 26,624 + 100,352 = 126,976
#define SMEM_BYTES (SA_BYTES + SH_BYTES)   // 126,976 -> 2 blocks / WGP

#define WMMA_BF16(A, B, C) \
    __builtin_amdgcn_wmma_f32_16x16x32_bf16(false, (A), false, (B), (short)0, (C), false, false)

// ---------------- prep kernels ----------------
__global__ void cast_bf16_k(const float* __restrict__ in, unsigned short* __restrict__ outp, long n) {
    long i = (long)blockIdx.x * blockDim.x + threadIdx.x;
    long stride = (long)gridDim.x * blockDim.x;
    for (; i < n; i += stride) outp[i] = f2bf(in[i]);
}

// in[b][r][c] (f32) -> out[b][c*rows + r] (bf16)
__global__ void tcast_k(const float* __restrict__ in, unsigned short* __restrict__ outp,
                        int batch, int rows, int cols) {
    long per = (long)rows * cols;
    long n = per * batch;
    long i = (long)blockIdx.x * blockDim.x + threadIdx.x;
    long stride = (long)gridDim.x * blockDim.x;
    for (; i < n; i += stride) {
        long b = i / per;
        long rem = i - b * per;
        int r = (int)(rem / cols);
        int c = (int)(rem - (long)r * cols);
        outp[b * per + (long)c * rows + r] = f2bf(in[i]);
    }
}

// ---------------- fused main kernel ----------------
__global__ __launch_bounds__(256) void fused_sparse_block(
    const float* __restrict__ xF,  const float* __restrict__ lnw,
    const float* __restrict__ lnb, const float* __restrict__ gmv,
    const int*   __restrict__ nbr,
    const unsigned short* __restrict__ xB,  const unsigned short* __restrict__ WT,
    const unsigned short* __restrict__ W1T, const unsigned short* __restrict__ W2T,
    const unsigned short* __restrict__ zpage,
    float* __restrict__ out)
{
    __shared__ __align__(16) char smem[SMEM_BYTES];

    const int tid    = threadIdx.x;
    const int lane   = tid & 31;
    const int wv     = tid >> 5;
    const int laneM  = lane & 15;
    const int laneHi = lane >> 4;
    const int rowW   = wv * 16;
    const long tileBase = (long)blockIdx.x * TM;

    // -------- async-gather descriptors (128 rows x 12 chunks = 1536 = 256*6) ----
    int gPart[6]; unsigned gLds[6]; long gPt[6]; bool gVal[6];
#pragma unroll
    for (int j = 0; j < 6; j++) {
        int id  = tid + 256 * j;
        int r   = id / 12;
        gPart[j] = id % 12;
        gLds[j]  = (unsigned)(unsigned long long)(const void*)
                       &((unsigned short*)smem)[r * SAS + gPart[j] * 8];   // buffer 0 offset
        gPt[j]   = tileBase + r;
        gVal[j]  = gPt[j] < N_PTS;
    }
    // weight stage: 96 rows x 12 chunks = 1152 chunks (waves 0-3: 5 chunks, waves 4-7: 4)
    int wRow[5], wPart[5]; unsigned wLds[5];
#pragma unroll
    for (int j = 0; j < 5; j++) {
        int id = tid + 256 * j;
        bool act = id < 1152;
        int r = act ? (id / 12) : 0;
        int p = act ? (id % 12) : 0;
        wRow[j] = r; wPart[j] = p;
        wLds[j] = (unsigned)(unsigned long long)(const void*)
                      &((unsigned short*)(smem + 2 * SA_BYTES))[r * SWS + p * 8];
    }

    // idx pipeline (2 iterations ahead)
    int idxC[6];
#pragma unroll
    for (int j = 0; j < 6; j++)
        idxC[j] = gVal[j] ? nbr[gPt[j]] : -1;                 // k = 0

    // issue helper (buf = 0/1)
    auto issue_async = [&](int k, int buf, const int* idxv) {
        unsigned aOff = (unsigned)(buf * SA_BYTES);
        unsigned wOff = (unsigned)(buf * SW_BYTES);
#pragma unroll
        for (int j = 0; j < 6; j++) {
            const unsigned short* src = (idxv[j] >= 0)
                ? (xB + (size_t)idxv[j] * CDIM + gPart[j] * 8)
                : (zpage + gPart[j] * 8);
            async_b128(gLds[j] + aOff, src);
        }
        const unsigned short* wk = WT + (size_t)k * (CDIM * CDIM);
#pragma unroll
        for (int j = 0; j < 4; j++)
            async_b128(wLds[j] + wOff, wk + wRow[j] * CDIM + wPart[j] * 8);
        if (tid < 128)                                        // wave-uniform (waves 0-3)
            async_b128(wLds[4] + wOff, wk + wRow[4] * CDIM + wPart[4] * 8);
    };

    // prologue: fill buffer 0 for k=0, preload idx for k=1
    issue_async(0, 0, idxC);
#pragma unroll
    for (int j = 0; j < 6; j++)
        idxC[j] = gVal[j] ? nbr[(size_t)N_PTS + gPt[j]] : -1; // k = 1

    v8f acc[6];
#pragma unroll
    for (int t = 0; t < 6; t++)
#pragma unroll
        for (int r = 0; r < 8; r++) acc[t][r] = 0.0f;

    // ================= sparse conv: 343 offsets, double-buffered =================
    for (int k = 0; k < KOFF; k++) {
        const int cur = k & 1;

        if (k < KOFF - 1) {
            // overlap: issue k+1 transfers into the other buffer
            issue_async(k + 1, cur ^ 1, idxC);
            // preload indices for k+2 (latency hidden behind compute)
            if (k + 2 < KOFF) {
#pragma unroll
                for (int j = 0; j < 6; j++)
                    idxC[j] = gVal[j] ? nbr[(size_t)(k + 2) * N_PTS + gPt[j]] : -1;
            }
            __builtin_prefetch(WT + (size_t)(k + 1) * (CDIM * CDIM) + tid * 36, 0, 1);
            // wait only for the PREVIOUS buffer's transfers (in-order completion)
            if (tid < 128) wait_async_le<11>(); else wait_async_le<10>();
        } else {
            wait_async_le<0>();
        }
        __syncthreads();

        // ---- 16x96 x 96x96 per wave : preloaded fragments, pipelined B ----
        {
            const unsigned short* sAb = (const unsigned short*)(smem + cur * SA_BYTES);
            const unsigned short* sWb = (const unsigned short*)(smem + 2 * SA_BYTES + cur * SW_BYTES);
            const unsigned short* pa0 = &sAb[(rowW + laneM) * SAS + laneHi * 16];
            const unsigned short* pb0 = &sWb[laneM * SWS + laneHi * 16];

            v16bf aF[3];
#pragma unroll
            for (int c = 0; c < 3; c++)
                aF[c] = make_frag(pa0 + c * 32, pa0 + c * 32 + 8);

            v16bf bF[6];
#pragma unroll
            for (int t = 0; t < 6; t++) {
                const unsigned short* pb = pb0 + t * 16 * SWS;
                bF[t] = make_frag(pb, pb + 8);
            }
#pragma unroll
            for (int c = 0; c < 3; c++) {
                v16bf bN[6];
                if (c < 2) {
#pragma unroll
                    for (int t = 0; t < 6; t++) {
                        const unsigned short* pb = pb0 + t * 16 * SWS + (c + 1) * 32;
                        bN[t] = make_frag(pb, pb + 8);
                    }
                }
#pragma unroll
                for (int t = 0; t < 6; t++)
                    acc[t] = WMMA_BF16(aF[c], bF[t], acc[t]);
                if (c < 2) {
#pragma unroll
                    for (int t = 0; t < 6; t++) bF[t] = bN[t];
                }
            }
        }
        __syncthreads();   // protect buffer reuse two iterations later
    }

    // ================= write conv result to LDS (f32) =================
    float*          sX  = (float*)(smem + SA_BYTES);          // reuses sA1/sW0 region
    unsigned short* sLN = (unsigned short*)smem;              // reuses sA0 region
    unsigned short* sH  = (unsigned short*)(smem + SA_BYTES); // reuses sX region (after LN)

#pragma unroll
    for (int t = 0; t < 6; t++)
#pragma unroll
        for (int r = 0; r < 8; r++)
            sX[(rowW + r + 8 * laneHi) * SXS + t * 16 + laneM] = acc[t][r];
    __syncthreads();

    // ================= LayerNorm (one thread per point) =================
    if (tid < TM) {
        const float* px = &sX[tid * SXS];
        float s = 0.0f;
#pragma unroll 16
        for (int c = 0; c < CDIM; c++) s += px[c];
        float mu = s * (1.0f / CDIM);
        float s2 = 0.0f;
#pragma unroll 16
        for (int c = 0; c < CDIM; c++) { float d = px[c] - mu; s2 += d * d; }
        float rs = rsqrtf(s2 * (1.0f / CDIM) + 1e-6f);
        unsigned short* pd = &sLN[tid * SAS];
#pragma unroll 16
        for (int c = 0; c < CDIM; c++)
            pd[c] = f2bf((px[c] - mu) * rs * lnw[c] + lnb[c]);
    }
    __syncthreads();

    // ================= MLP GEMM1 (96 -> 384) + exact GELU =================
    v16bf a1[3];
#pragma unroll
    for (int c = 0; c < 3; c++) {
        const unsigned short* pa = &sLN[(rowW + laneM) * SAS + c * 32 + laneHi * 16];
        a1[c] = make_frag(pa, pa + 8);
    }
#pragma unroll
    for (int g = 0; g < 4; g++) {
        v8f h[6];
#pragma unroll
        for (int t = 0; t < 6; t++)
#pragma unroll
            for (int r = 0; r < 8; r++) h[t][r] = 0.0f;
#pragma unroll
        for (int c = 0; c < 3; c++) {
#pragma unroll
            for (int t = 0; t < 6; t++) {
                const unsigned short* pb =
                    &W1T[(size_t)(g * 96 + t * 16 + laneM) * CDIM + c * 32 + laneHi * 16];
                v16bf b = make_frag(pb, pb + 8);
                h[t] = WMMA_BF16(a1[c], b, h[t]);
            }
        }
#pragma unroll
        for (int t = 0; t < 6; t++)
#pragma unroll
            for (int r = 0; r < 8; r++) {
                float x  = h[t][r];
                float ge = 0.5f * x * (1.0f + erff(x * 0.70710678118654752f));
                sH[(rowW + r + 8 * laneHi) * SHS + g * 96 + t * 16 + laneM] = f2bf(ge);
            }
    }
    // no barrier: each wave consumes only the sH rows it wrote (per-wave LDS ordering)

    // ================= MLP GEMM2 (384 -> 96) =================
    v8f o[6];
#pragma unroll
    for (int t = 0; t < 6; t++)
#pragma unroll
        for (int r = 0; r < 8; r++) o[t][r] = 0.0f;
#pragma unroll
    for (int kc = 0; kc < 12; kc++) {
        const unsigned short* pa = &sH[(rowW + laneM) * SHS + kc * 32 + laneHi * 16];
        v16bf a = make_frag(pa, pa + 8);
#pragma unroll
        for (int t = 0; t < 6; t++) {
            const unsigned short* pb =
                &W2T[(size_t)(t * 16 + laneM) * HID + kc * 32 + laneHi * 16];
            v16bf b = make_frag(pb, pb + 8);
            o[t] = WMMA_BF16(a, b, o[t]);
        }
    }

    // ================= epilogue: out = xF + gamma * mlp =================
#pragma unroll
    for (int t = 0; t < 6; t++) {
        int col = t * 16 + laneM;
        float gsc = gmv[col];
#pragma unroll
        for (int r = 0; r < 8; r++) {
            long p = tileBase + rowW + r + 8 * laneHi;
            if (p < N_PTS)
                out[p * CDIM + col] = xF[p * CDIM + col] + gsc * o[t][r];
        }
    }
}

// ---------------- host launcher ----------------
extern "C" void kernel_launch(void* const* d_in, const int* in_sizes, int n_in,
                              void* d_out, int out_size, void* d_ws, size_t ws_size,
                              hipStream_t stream) {
    const float* xF  = (const float*)d_in[0];
    const float* Wc  = (const float*)d_in[1];
    const float* lnw = (const float*)d_in[2];
    const float* lnb = (const float*)d_in[3];
    const float* W1  = (const float*)d_in[4];
    const float* W2  = (const float*)d_in[5];
    const float* gm  = (const float*)d_in[6];
    const int*   nbr = (const int*)d_in[7];
    float* out = (float*)d_out;

    char* ws = (char*)d_ws;
    size_t off = 0;
    unsigned short* xB  = (unsigned short*)(ws + off); off += (size_t)N_PTS * CDIM * 2;        // 38,400,000
    unsigned short* WT  = (unsigned short*)(ws + off); off += (size_t)KOFF * CDIM * CDIM * 2;  //  6,322,176
    unsigned short* W1T = (unsigned short*)(ws + off); off += (size_t)CDIM * HID * 2;          //     73,728
    unsigned short* W2T = (unsigned short*)(ws + off); off += (size_t)HID * CDIM * 2;          //     73,728
    unsigned short* zp  = (unsigned short*)(ws + off); off += 256;                             // zero page

    hipMemsetAsync(zp, 0, 256, stream);
    cast_bf16_k<<<2048, 256, 0, stream>>>(xF, xB, (long)N_PTS * CDIM);
    tcast_k<<<2048, 256, 0, stream>>>(Wc, WT, KOFF, CDIM, CDIM);
    tcast_k<<<96, 256, 0, stream>>>(W1, W1T, 1, CDIM, HID);
    tcast_k<<<96, 256, 0, stream>>>(W2, W2T, 1, HID, CDIM);
    fused_sparse_block<<<NT, 256, 0, stream>>>(xF, lnw, lnb, gm, nbr, xB, WT, W1T, W2T, zp, out);
}
